// GaPartNetWithFlows_24163486007685
// MI455X (gfx1250) — 8-bit matrix operand order repaired and checked by reading, compiled, not run on hardware
//
#include <hip/hip_runtime.h>
#include <hip/hip_bf16.h>


#define P 512
#define NPTS 100000
#define KPAD 100352
#define KSPLIT 8
#define KCH (KPAD / KSPLIT)
#define A_BYTES ((size_t)P * (size_t)KPAD)

typedef int v8i __attribute__((ext_vector_type(8)));

__global__ void zero_ws_kernel(uint4* __restrict__ p, size_t n4) {
  size_t i = (size_t)blockIdx.x * blockDim.x + threadIdx.x;
  size_t stride = (size_t)gridDim.x * blockDim.x;
  uint4 z = {0u, 0u, 0u, 0u};
  for (; i < n4; i += stride) p[i] = z;
}

__global__ void scatter_kernel(const int* __restrict__ sorted_indices,
                               const int* __restrict__ proposal_indices,
                               unsigned* __restrict__ A32, int T) {
  int t = blockIdx.x * blockDim.x + threadIdx.x;
  if (t >= T) return;
  unsigned idx = (unsigned)proposal_indices[t] * (unsigned)KPAD +
                 (unsigned)sorted_indices[t];
  atomicAdd(&A32[idx >> 2], 1u << ((idx & 3u) * 8u));
}

__global__ void __launch_bounds__(256)
gemm_iu8_kernel(const unsigned char* __restrict__ A, int* __restrict__ inter) {
  const int lane = threadIdx.x & 31;
  const int wave = threadIdx.x >> 5;
  const int job = blockIdx.x * 8 + wave;
  const int macro = job >> 3;
  const int ks = job & (KSPLIT - 1);
  const int tm = macro >> 4;
  const int tn = macro & 15;
  const int m = lane & 15;
  const int hi = lane >> 4;

  const unsigned char* rowA0 = A + (size_t)(tm * 32 + m) * KPAD;
  const unsigned char* rowA1 = rowA0 + (size_t)16 * KPAD;
  const unsigned char* rowB0 = A + (size_t)(tn * 32 + m) * KPAD;
  const unsigned char* rowB1 = rowB0 + (size_t)16 * KPAD;

  int koffA[8], koffB[8];
#pragma unroll
  for (int v = 0; v < 8; ++v) {
    koffA[v] = 16 * ((v >> 1) & 1) + 32 * (v >> 2) + 4 * (v & 1) + 8 * hi;
    koffB[v] = 16 * ((v >> 1) & 1) + 32 * (v >> 2) + 4 * (v & 1) + 8 * hi;
  }

  v8i c00 = {}, c01 = {}, c10 = {}, c11 = {};
  const int k0 = ks * KCH;
  const int k1 = k0 + KCH;
  for (int k = k0; k < k1; k += 64) {
    v8i a0, a1, b0, b1;
#pragma unroll
    for (int v = 0; v < 8; ++v) {
      a0[v] = *(const int*)(rowA0 + k + koffA[v]);
      a1[v] = *(const int*)(rowA1 + k + koffA[v]);
      b0[v] = *(const int*)(rowB0 + k + koffB[v]);
      b1[v] = *(const int*)(rowB1 + k + koffB[v]);
    }
    __builtin_prefetch(rowA0 + k + 512, 0, 1);
    __builtin_prefetch(rowA1 + k + 512, 0, 1);
    __builtin_prefetch(rowB0 + k + 512, 0, 1);
    __builtin_prefetch(rowB1 + k + 512, 0, 1);
    c00 = __builtin_amdgcn_wmma_i32_16x16x64_iu8(false, a0, false, b0, c00,
                                                 false, false);
    c01 = __builtin_amdgcn_wmma_i32_16x16x64_iu8(false, a0, false, b1, c01,
                                                 false, false);
    c10 = __builtin_amdgcn_wmma_i32_16x16x64_iu8(false, a1, false, b0, c10,
                                                 false, false);
    c11 = __builtin_amdgcn_wmma_i32_16x16x64_iu8(false, a1, false, b1, c11,
                                                 false, false);
  }

  const int Mb = tm * 32 + 8 * hi;
  const int Nb = tn * 32 + (lane & 15);
#pragma unroll
  for (int r = 0; r < 8; ++r) {
    atomicAdd(&inter[(Mb + r) * P + Nb], c00[r]);
    atomicAdd(&inter[(Mb + r) * P + Nb + 16], c01[r]);
    atomicAdd(&inter[(Mb + r + 16) * P + Nb], c10[r]);
    atomicAdd(&inter[(Mb + r + 16) * P + Nb + 16], c11[r]);
  }
}

__global__ void rowsum_kernel(const unsigned* __restrict__ A32,
                              int* __restrict__ nOut) {
  const int row = blockIdx.x;
  const unsigned* r = A32 + (size_t)row * (KPAD / 4);
  int s = 0;
  for (int i = threadIdx.x; i < KPAD / 4; i += 256) {
    unsigned x = r[i];
    s += (int)((x & 255u) + ((x >> 8) & 255u) + ((x >> 16) & 255u) + (x >> 24));
  }
  __shared__ int red[256];
  red[threadIdx.x] = s;
  __syncthreads();
  for (int off = 128; off > 0; off >>= 1) {
    if (threadIdx.x < off) red[threadIdx.x] += red[threadIdx.x + off];
    __syncthreads();
  }
  if (threadIdx.x == 0) nOut[row] = red[0];
}

__global__ void iou_kernel(const int* __restrict__ inter,
                           const int* __restrict__ n,
                           float* __restrict__ ious) {
  int idx = blockIdx.x * 256 + threadIdx.x;
  int i = idx >> 9;
  int j = idx & (P - 1);
  float I = (float)inter[idx];
  float U = (float)(n[i] + n[j]) - I;
  ious[idx] = I / (U + 1e-8f);
}

__global__ void nms_kernel(const float* __restrict__ ious,
                           const float* __restrict__ scores,
                           float* __restrict__ keep_out) {
  __shared__ float sc[P];
  __shared__ int order[P];
  __shared__ int keep_s[P];
  const int tid = threadIdx.x;

  sc[tid] = scores[tid];
  __syncthreads();

  float s = sc[tid];
  int rank = 0;
  for (int j = 0; j < P; ++j) {
    float o = sc[j];
    rank += (o > s) || (o == s && j < tid);
  }
  order[rank] = tid;
  keep_s[tid] = 1;
  __syncthreads();

  const int myOrig = order[tid];
  for (int i = 0; i < P - 1; ++i) {
    __syncthreads();
    if (keep_s[i] && tid > i) {
      float v = ious[order[i] * P + myOrig];
      if (v > 0.3f) keep_s[tid] = 0;
    }
  }
  __syncthreads();
  keep_out[myOrig] = (float)keep_s[tid];
}

extern "C" void kernel_launch(void* const* d_in, const int* in_sizes, int n_in,
                              void* d_out, int out_size, void* d_ws,
                              size_t ws_size, hipStream_t stream) {
  const int* sorted_indices = (const int*)d_in[0];
  const int* proposal_indices = (const int*)d_in[1];
  const float* score_preds = (const float*)d_in[3];
  const int T = in_sizes[0];

  unsigned char* A = (unsigned char*)d_ws;
  int* inter = (int*)((char*)d_ws + A_BYTES);
  int* nrow = (int*)((char*)d_ws + A_BYTES + (size_t)P * P * 4);

  float* ious_out = (float*)d_out;
  float* keep_out = (float*)d_out + (size_t)P * P;

  const size_t zero_bytes = A_BYTES + (size_t)P * P * 4;
  zero_ws_kernel<<<2048, 256, 0, stream>>>((uint4*)A, zero_bytes / 16);
  scatter_kernel<<<(T + 255) / 256, 256, 0, stream>>>(
      sorted_indices, proposal_indices, (unsigned*)A, T);
  gemm_iu8_kernel<<<256, 256, 0, stream>>>(A, inter);
  rowsum_kernel<<<P, 256, 0, stream>>>((const unsigned*)A, nrow);
  iou_kernel<<<(P * P) / 256, 256, 0, stream>>>(inter, nrow, ious_out);
  nms_kernel<<<1, P, 0, stream>>>(ious_out, score_preds, keep_out);
}
